// Deformation_49624052138549
// MI455X (gfx1250) — compile-verified, matrix operand order checked by reading, not run on hardware
//
#include <hip/hip_runtime.h>
#include <hip/hip_bf16.h>

// ---------------- problem constants (match reference) ----------------
constexpr int N0 = 10000, P1 = 30000, N1 = 40000, P2 = 120000, N2 = 160000;
constexpr int E0 = 60000, E1 = 120000, E2 = 480000;
constexpr int IN = 3, H = 192, OUT = 3;
constexpr int KC = 576;        // (K+1)*H  — GEMM contraction dim for mid layers
constexpr int KT = KC / 32;    // 18 k-tiles of 32
constexpr int NT = H / 16;     // 12 n-tiles of 16
constexpr int FRAG = 512;      // halves per 32x16 B fragment (32 lanes x 16)
constexpr int LSTRIDE = KT * NT * FRAG;   // packed bf16 halves per layer
constexpr int MT = 4;          // M-tiles (of 16 rows) per block -> 64 rows/block

typedef __bf16 bf16;
typedef bf16  v16bf __attribute__((ext_vector_type(16)));
typedef float v8f   __attribute__((ext_vector_type(8)));

// ---------------- weight pre-pack: fp32 [L][576][192] -> bf16 fragments ----------------
__global__ void pack_w_kernel(const float* __restrict__ Wmid, bf16* __restrict__ Wp, int total)
{
    int idx = blockIdx.x * blockDim.x + threadIdx.x;
    if (idx >= total) return;
    int e    = idx & 15;
    int lane = (idx >> 4) & 31;
    int nt   = (idx >> 9) % NT;
    int kt   = (idx / (FRAG * NT)) % KT;
    int L    = idx / (FRAG * NT * KT);
    // K striping identical to the 16-bit A-matrix VGPR layout (ISA 7.12.2)
    int k = kt * 32 + ((e >> 3) << 4) + ((lane >> 4) << 3) + (e & 7);
    int n = nt * 16 + (lane & 15);
    Wp[idx] = (bf16)Wmid[(size_t)L * KC * H + (size_t)k * H + n];
}

// ---------------- graph helpers ----------------
__device__ __forceinline__ void edge_sd(const int* src, const int* dst, const int* pair,
                                        int E, int e, int& s, int& d)
{
    if (pair) {
        int e2 = (e < E) ? e : e - E;
        int a = pair[2 * e2], b = pair[2 * e2 + 1];
        s = (e < E) ? a : b;
        d = (e < E) ? b : a;
    } else {
        s = src[e]; d = dst[e];
    }
}

__global__ void degree_kernel(const int* src, const int* dst, const int* pair,
                              int E, int bidir, float* __restrict__ deg)
{
    int t = blockIdx.x * blockDim.x + threadIdx.x;
    int Et = bidir ? 2 * E : E;
    if (t >= Et) return;
    int s, d;
    edge_sd(src, dst, pair, E, t, s, d);
    atomicAdd(&deg[d], 1.0f);
}

__global__ void norm_kernel(float* __restrict__ deg, int n)
{
    int i = blockIdx.x * blockDim.x + threadIdx.x;
    if (i < n) deg[i] = rsqrtf(fmaxf(deg[i], 1.0f));  // in-place: deg -> D^{-1/2}
}

// out[dst, f] += x[src, f] * norm[src]   (edge x feature parallel)
__global__ void scatter_kernel(const float* __restrict__ x, const float* __restrict__ nrm,
                               const int* src, const int* dst, const int* pair,
                               int E, int bidir, float* __restrict__ out, int F)
{
    long long t = (long long)blockIdx.x * blockDim.x + threadIdx.x;
    long long Et = (long long)(bidir ? 2 * E : E) * F;
    if (t >= Et) return;
    int e = (int)(t / F);
    int f = (int)(t - (long long)e * F);
    int s, d;
    edge_sd(src, dst, pair, E, e, s, d);
    atomicAdd(&out[(size_t)d * F + f], x[(size_t)s * F + f] * nrm[s]);
}

__global__ void scale_kernel(float* __restrict__ h, const float* __restrict__ nrm,
                             int n, int F)
{
    long long t = (long long)blockIdx.x * blockDim.x + threadIdx.x;
    if (t >= (long long)n * F) return;
    h[t] *= nrm[t / F];
}

// ---------------- skinny layers (scalar) ----------------
__global__ void first_gemm_kernel(const float* __restrict__ x3, const float* __restrict__ h1,
                                  const float* __restrict__ h2,
                                  const float* __restrict__ Wf, const float* __restrict__ bf_,
                                  float* __restrict__ out, int n)
{
    int t = blockIdx.x * blockDim.x + threadIdx.x;
    if (t >= n * H) return;
    int i = t / H, o = t - i * H;
    float acc = bf_[o];
#pragma unroll
    for (int k = 0; k < 3; ++k) acc += x3[i * 3 + k] * Wf[k * H + o];
#pragma unroll
    for (int k = 0; k < 3; ++k) acc += h1[i * 3 + k] * Wf[(3 + k) * H + o];
#pragma unroll
    for (int k = 0; k < 3; ++k) acc += h2[i * 3 + k] * Wf[(6 + k) * H + o];
    out[t] = fmaxf(acc, 0.0f);
}

__global__ void last_gemm_kernel(const float* __restrict__ x, const float* __restrict__ h1,
                                 const float* __restrict__ h2,
                                 const float* __restrict__ Wl, const float* __restrict__ bl,
                                 float* __restrict__ out, int n)
{
    int t = blockIdx.x * blockDim.x + threadIdx.x;
    if (t >= n * OUT) return;
    int i = t / OUT, o = t - i * OUT;
    const float* xr  = x  + (size_t)i * H;
    const float* h1r = h1 + (size_t)i * H;
    const float* h2r = h2 + (size_t)i * H;
    float acc = bl[o];
#pragma unroll 4
    for (int k = 0; k < H; ++k) {
        acc += xr[k]  * Wl[k * OUT + o];
        acc += h1r[k] * Wl[(H + k) * OUT + o];
        acc += h2r[k] * Wl[(2 * H + k) * OUT + o];
    }
    out[t] = acc;  // identity activation
}

// hx = 0.5*(prev + cur); safe in-place when hx aliases prev (elementwise)
__global__ void avg_kernel(float* __restrict__ hx, const float* __restrict__ prev,
                           const float* __restrict__ cur, long long n)
{
    long long t = (long long)blockIdx.x * blockDim.x + threadIdx.x;
    if (t < n) hx[t] = 0.5f * (prev[t] + cur[t]);
}

__global__ void unpool_kernel(const float* __restrict__ feat, const int* __restrict__ pidx,
                              float* __restrict__ dst_, int n, int P)
{
    int t = blockIdx.x * blockDim.x + threadIdx.x;
    int tot = (n + P) * 3;
    if (t >= tot) return;
    int i = t / 3, f = t - i * 3;
    float v;
    if (i < n) v = feat[i * 3 + f];
    else {
        int j = i - n;
        int a = pidx[2 * j], b = pidx[2 * j + 1];
        v = 0.5f * (feat[a * 3 + f] + feat[b * 3 + f]);
    }
    dst_[t] = v;
}

// ---------------- mid-layer GEMM: [n,576] @ [576,192] via v_wmma_f32_16x16x32_bf16 ----------------
// Block: 128 threads = 4 waves; block owns a 64-row M super-tile (4 x 16-row M-tiles in LDS).
// Wave w owns N-tiles {3w,3w+1,3w+2} across ALL 4 M-tiles: 12 independent WMMA chains
// (96 accumulator VGPRs), and each B fragment loaded once per k-step is reused 4x from
// registers -> 4x less L2 weight traffic per output row and deep WMMA latency hiding.
__global__ __launch_bounds__(128, 1)
void mid_gemm_wmma(const float* __restrict__ x, const float* __restrict__ h1,
                   const float* __restrict__ h2,
                   const bf16* __restrict__ Wp, const float* __restrict__ bias,
                   float* __restrict__ out, int n)
{
    __shared__ bf16 sA[MT * KT * 32 * 16];   // 4 M-tiles in fragment order, 73728 B

    const int tid  = threadIdx.x;
    const int lane = tid & 31;
    const int wave = tid >> 5;          // 0..3
    const int m0   = blockIdx.x * (16 * MT);

    // ---- stage A (64 rows x 576 cols, fp32 from x|h1|h2) into LDS as bf16 fragments ----
    for (int p = tid; p < MT * KT * 32; p += 128) {
        int mt  = p / (KT * 32);
        int q   = p - mt * (KT * 32);
        int kt  = q >> 5;
        int ln  = q & 31;
        int row = m0 + mt * 16 + (ln & 15);
        int l8  = (ln >> 4) << 3;       // 0 or 8
        int kb  = kt * 32 + l8;         // start of first run of 8
        v16bf frag;
        if (row < n) {
            int seg = kt / 6;           // 192-col segment: 0->x, 1->h1, 2->h2
            const float* s_ = (seg == 0 ? x : (seg == 1 ? h1 : h2))
                              + (size_t)row * H + (kb - seg * H);
#pragma unroll
            for (int e = 0; e < 8; ++e) frag[e] = (bf16)s_[e];
#pragma unroll
            for (int e = 0; e < 8; ++e) frag[8 + e] = (bf16)s_[16 + e];
        } else {
#pragma unroll
            for (int e = 0; e < 16; ++e) frag[e] = (bf16)0.0f;
        }
        *(v16bf*)(&sA[p * 16]) = frag;  // 32-byte LDS store
    }
    __syncthreads();

    // ---- 12 accumulators [mt][j], initialized with bias ----
    const int cb = (lane & 15);
    v8f acc[MT][3];
#pragma unroll
    for (int mt = 0; mt < MT; ++mt)
#pragma unroll
        for (int j = 0; j < 3; ++j) {
            float bv = bias[(wave * 3 + j) * 16 + cb];
#pragma unroll
            for (int v = 0; v < 8; ++v) acc[mt][j][v] = bv;
        }

    // ---- main K loop: per kt -> 4 LDS A-frags + 3 global B-frags -> 12 WMMAs ----
#pragma unroll 1
    for (int kt = 0; kt < KT; ++kt) {
        const bf16* wb = Wp + ((size_t)kt * NT + wave * 3) * FRAG + lane * 16;
        v16bf b0 = *(const v16bf*)(wb);
        v16bf b1 = *(const v16bf*)(wb + FRAG);
        v16bf b2 = *(const v16bf*)(wb + 2 * FRAG);
#pragma unroll
        for (int mt = 0; mt < MT; ++mt) {
            v16bf a = *(const v16bf*)(&sA[((mt * KT + kt) * 32 + lane) * 16]);
            acc[mt][0] = __builtin_amdgcn_wmma_f32_16x16x32_bf16(false, a, false, b0, (short)0, acc[mt][0], false, false);
            acc[mt][1] = __builtin_amdgcn_wmma_f32_16x16x32_bf16(false, a, false, b1, (short)0, acc[mt][1], false, false);
            acc[mt][2] = __builtin_amdgcn_wmma_f32_16x16x32_bf16(false, a, false, b2, (short)0, acc[mt][2], false, false);
        }
    }

    // ---- writeback with fused ReLU (C/D layout: lane<16 -> M=v, lane>=16 -> M=8+v) ----
    const int c0 = (wave * 3) * 16 + cb;
#pragma unroll
    for (int mt = 0; mt < MT; ++mt) {
#pragma unroll
        for (int v = 0; v < 8; ++v) {
            int row = m0 + mt * 16 + v + ((lane >> 4) << 3);
            if (row < n) {
                float* o = out + (size_t)row * H + c0;
                o[0]  = fmaxf(acc[mt][0][v], 0.0f);
                o[16] = fmaxf(acc[mt][1][v], 0.0f);
                o[32] = fmaxf(acc[mt][2][v], 0.0f);
            }
        }
    }
}

// ---------------- host orchestration ----------------
static inline int cdiv(long long a, int b) { return (int)((a + b - 1) / b); }

static void run_stage(int s, int n,
                      const int* src, const int* dst, const int* pair, int E, int bidir,
                      const float* x3, float* outp,
                      float* nrm, float* h1, float* h2,
                      float* bufA, float* bufB, float* bufC,
                      const bf16* Wpack,
                      const float* Wf, const float* bf_,
                      const float* bm, const float* Wl, const float* bl,
                      hipStream_t stream)
{
    const int Et = bidir ? 2 * E : E;

    // norm = D^{-1/2}
    hipMemsetAsync(nrm, 0, (size_t)n * 4, stream);
    degree_kernel<<<cdiv(Et, 256), 256, 0, stream>>>(src, dst, pair, E, bidir, nrm);
    norm_kernel<<<cdiv(n, 256), 256, 0, stream>>>(nrm, n);

    auto hop = [&](const float* in, float* outh, int F) {
        hipMemsetAsync(outh, 0, (size_t)n * F * 4, stream);
        scatter_kernel<<<cdiv((long long)Et * F, 256), 256, 0, stream>>>(
            in, nrm, src, dst, pair, E, bidir, outh, F);
        scale_kernel<<<cdiv((long long)n * F, 256), 256, 0, stream>>>(outh, nrm, n, F);
    };

    // first layer (F=3 aggregation, 9->192 scalar GEMM + ReLU)
    hop(x3, h1, 3);
    hop(h1, h2, 3);
    first_gemm_kernel<<<cdiv((long long)n * H, 256), 256, 0, stream>>>(
        x3, h1, h2, Wf + (size_t)s * 9 * H, bf_ + (size_t)s * H, bufA, n);

    float* xp = bufA;   // current layer input
    float* pp = bufA;   // prev (raw relu output of previous layer)

    for (int l = 0; l < 12; ++l) {
        float* cur = (bufA != xp && bufA != pp) ? bufA
                   : (bufB != xp && bufB != pp) ? bufB : bufC;
        hop(xp, h1, H);
        hop(h1, h2, H);
        mid_gemm_wmma<<<cdiv(n, 16 * MT), 128, 0, stream>>>(
            xp, h1, h2,
            Wpack + (size_t)(s * 12 + l) * LSTRIDE,
            bm + (size_t)(s * 12 + l) * H, cur, n);
        if (l & 1) {   // reference: after layers where (i+1) is even, h = 0.5*(prev+cur)
            avg_kernel<<<cdiv((long long)n * H, 256), 256, 0, stream>>>(
                xp, pp, cur, (long long)n * H);   // write into old-x buffer (in-place safe)
            pp = cur;  // xp stays (now holds the average)
        } else {
            xp = cur; pp = cur;
        }
    }

    // last layer (576 -> 3, identity activation)
    hop(xp, h1, H);
    hop(h1, h2, H);
    last_gemm_kernel<<<cdiv((long long)n * OUT, 256), 256, 0, stream>>>(
        xp, h1, h2, Wl + (size_t)s * KC * OUT, bl + (size_t)s * OUT, outp, n);
}

extern "C" void kernel_launch(void* const* d_in, const int* in_sizes, int n_in,
                              void* d_out, int out_size, void* d_ws, size_t ws_size,
                              hipStream_t stream)
{
    (void)in_sizes; (void)n_in; (void)out_size; (void)ws_size;

    const float* features = (const float*)d_in[0];
    const int*   src0     = (const int*)d_in[1];
    const int*   dst0     = (const int*)d_in[2];
    const int*   pool0    = (const int*)d_in[3];
    const int*   edge0    = (const int*)d_in[4];
    const int*   pool1    = (const int*)d_in[5];
    const int*   edge1    = (const int*)d_in[6];
    const float* Wf       = (const float*)d_in[7];
    const float* bf_      = (const float*)d_in[8];
    const float* Wm       = (const float*)d_in[9];
    const float* bm       = (const float*)d_in[10];
    const float* Wl       = (const float*)d_in[11];
    const float* bl       = (const float*)d_in[12];
    float* out = (float*)d_out;

    // d_out layout: out1[N0*3] | out2[N1*3] | out3[N2*3] | out1_2[N1*3] | out2_2[N2*3]
    const size_t O2  = (size_t)N0 * 3;
    const size_t O3  = O2 + (size_t)N1 * 3;
    const size_t O12 = O3 + (size_t)N2 * 3;
    const size_t O22 = O12 + (size_t)N1 * 3;

    // workspace carve-up
    char* ws = (char*)d_ws;
    size_t off = 0;
    auto carve = [&](size_t bytes) -> char* {
        char* p = ws + off;
        off = (off + bytes + 255) & ~(size_t)255;
        return p;
    };
    float* nrm  = (float*)carve((size_t)N2 * 4);
    float* h1   = (float*)carve((size_t)N2 * H * 4);
    float* h2   = (float*)carve((size_t)N2 * H * 4);
    float* bufA = (float*)carve((size_t)N2 * H * 4);
    float* bufB = (float*)carve((size_t)N2 * H * 4);
    float* bufC = (float*)carve((size_t)N2 * H * 4);
    bf16*  Wpk  = (bf16*)carve((size_t)3 * 12 * LSTRIDE * sizeof(bf16));

    // pre-pack all mid-layer weights into WMMA fragment order (bf16)
    {
        int total = 3 * 12 * LSTRIDE;
        pack_w_kernel<<<cdiv(total, 256), 256, 0, stream>>>(Wm, Wpk, total);
    }

    // stage 1: directed edge lists src0/dst0
    run_stage(0, N0, src0, dst0, nullptr, E0, 0,
              features, out /*out1*/,
              nrm, h1, h2, bufA, bufB, bufC, Wpk, Wf, bf_, bm, Wl, bl, stream);

    unpool_kernel<<<cdiv((long long)N1 * 3, 256), 256, 0, stream>>>(
        out, pool0, out + O12, N0, P1);                     // out1_2 (also stage-2 input)

    // stage 2: bidirectional edge0
    run_stage(1, N1, nullptr, nullptr, edge0, E1, 1,
              out + O12, out + O2 /*out2*/,
              nrm, h1, h2, bufA, bufB, bufC, Wpk, Wf, bf_, bm, Wl, bl, stream);

    unpool_kernel<<<cdiv((long long)N2 * 3, 256), 256, 0, stream>>>(
        out + O2, pool1, out + O22, N1, P2);                // out2_2 (also stage-3 input)

    // stage 3: bidirectional edge1
    run_stage(2, N2, nullptr, nullptr, edge1, E2, 1,
              out + O22, out + O3 /*out3*/,
              nrm, h1, h2, bufA, bufB, bufC, Wpk, Wf, bf_, bm, Wl, bl, stream);
}
